// ConditionedPNA_68951404970531
// MI455X (gfx1250) — compile-verified, matrix operand order checked by reading, not run on hardware
//
#include <hip/hip_runtime.h>
#include <hip/hip_bf16.h>
#include <math.h>

// ---------------------------------------------------------------------------
// ConditionedPNA for MI455X (gfx1250).
//
// m_e = u[col] + v[row] + bpre with u = cur@Wpre_top, v = cur@Wpre_bot;
// all PNA aggregates decompose into per-node algebra over edge reductions of
// v[row].  GEMMs: v_wmma_f32_16x16x32_f16 (fp32 accumulate), weights
// pre-packed into fragment-ordered f16 so B operands are two b128 loads.
// Working set (~45 MB) is L2-resident (192 MB L2).
// ---------------------------------------------------------------------------

typedef __attribute__((ext_vector_type(16))) _Float16 v16h;
typedef __attribute__((ext_vector_type(8)))  _Float16 v8h;
typedef __attribute__((ext_vector_type(8)))  float    v8f;

#define FD    128
#define NN    8000          // B * N_NODES
#define NPB   2000          // N_NODES
#define NE    64000
#define ET    256000        // B * NE
#define NSORT 8192
#define KTOP  800

__device__ __forceinline__ void atomicMaxF(float* a, float v) {
    if (v >= 0.0f) atomicMax((int*)a, __float_as_int(v));
    else           atomicMin((unsigned int*)a, __float_as_uint(v));
}
__device__ __forceinline__ void atomicMinF(float* a, float v) {
    if (v >= 0.0f) atomicMin((int*)a, __float_as_int(v));
    else           atomicMax((unsigned int*)a, __float_as_uint(v));
}

// ---------------------------------------------------------------------------
// Pack W (K x 128 f32, row-major) into fragment-ordered f16 so that wave w,
// lane L, chunk kc reads its whole 16-half B operand contiguously:
//   Wpk[(((kc*128 + col)*2 + g)*16 + i)] = (f16) W[(kc*32 + kf(g,i))*128 + col]
// with col = w*16 + (L&15), g = L>>4, kf = g*8 + (i<8 ? i : i+8).
// ---------------------------------------------------------------------------
__global__ void pack_w_kernel(const float* __restrict__ W, _Float16* __restrict__ Wpk,
                              int K)
{
    int t = blockIdx.x * blockDim.x + threadIdx.x;
    if (t >= K * FD) return;
    int i   = t & 15;
    int g   = (t >> 4) & 1;
    int col = (t >> 5) & 127;
    int kc  = t >> 12;
    int kf  = g * 8 + (i < 8 ? i : i + 8);
    Wpk[t] = (_Float16)W[(size_t)(kc * 32 + kf) * FD + col];
}

// ---------------------------------------------------------------------------
// M x K @ K x 128 GEMM via WMMA f16, templated on all mode flags so the
// K-chunk loop is branch-free.  One workgroup = 16 rows of C; 8 waves each
// own a 16-wide column block.
// A element (m,k) = k < K0 ? A0[m*K0+k] : A1[(m/a1_div)*(K-K0)+(k-K0)]
// ---------------------------------------------------------------------------
template<bool HAS_A1, bool HAS_RS, bool HAS_EM, bool RELU, bool ACCUM>
__global__ __launch_bounds__(256)
void gemm_wmma_kernel(const float* __restrict__ A0, int K0,
                      const float* __restrict__ A1, int a1_div,
                      int K,
                      const _Float16* __restrict__ Wpk,   // packed, see above
                      const float* __restrict__ bias,     // 128 or null
                      const float* __restrict__ row_scale,// M (if HAS_RS)
                      const float* __restrict__ emul,     // M x 128 (if HAS_EM)
                      float* __restrict__ out)            // M x 128
{
    __shared__ alignas(32) _Float16 As[16 * 32];
    const int tid  = threadIdx.x;
    const int lane = tid & 31;
    const int wave = tid >> 5;          // 0..7 -> 16-col block
    const int m0   = blockIdx.x * 16;
    const int g    = lane >> 4;         // half-wave group
    const int mr   = lane & 15;
    const int col  = wave * 16 + mr;

    // staging coordinates (2 consecutive k per thread, same row)
    const int sidx = tid * 2;
    const int sr   = sidx >> 5;
    const int skk  = sidx & 31;
    const int sm   = m0 + sr;

    v8f c = {0.f,0.f,0.f,0.f,0.f,0.f,0.f,0.f};
    const int nCh = K >> 5;             // K is a multiple of 32 here

    for (int kc = 0; kc < nCh; ++kc) {
        __syncthreads();
        {   // stage 16x32 fp32 tile of A as f16
            int k = kc * 32 + skk;
            float2 v2;
            if constexpr (HAS_A1) {
                if (k < K0) v2 = *(const float2*)(A0 + (size_t)sm * K0 + k);
                else        v2 = *(const float2*)(A1 + (size_t)(sm / a1_div) * (K - K0)
                                                     + (k - K0));
            } else {
                v2 = *(const float2*)(A0 + (size_t)sm * K0 + k);
                if (kc + 1 < nCh)      // next-chunk prefetch -> global_prefetch_b8
                    __builtin_prefetch(A0 + (size_t)sm * K0 + k + 32, 0, 1);
            }
            if constexpr (HAS_RS) {
                float s = row_scale[sm];
                v2.x *= s; v2.y *= s;
            }
            As[sr * 32 + skk]     = (_Float16)v2.x;
            As[sr * 32 + skk + 1] = (_Float16)v2.y;
        }
        __syncthreads();

        // A fragment: two contiguous 16B LDS reads per lane
        v8h alo = *(const v8h*)(As + mr * 32 + g * 8);
        v8h ahi = *(const v8h*)(As + mr * 32 + g * 8 + 16);
        v16h a = __builtin_shufflevector(alo, ahi,
                    0,1,2,3,4,5,6,7,8,9,10,11,12,13,14,15);
        // B fragment: 32 contiguous bytes of pre-packed f16 (two b128 loads)
        v16h b = *(const v16h*)(Wpk + (((size_t)kc * FD + col) * 2 + g) * 16);

        c = __builtin_amdgcn_wmma_f32_16x16x32_f16(
                false, a, false, b, (short)0, c, false, false);
    }

    const float bcol = bias ? bias[col] : 0.0f;
    #pragma unroll
    for (int r = 0; r < 8; ++r) {
        int m = m0 + g * 8 + r;
        float v = c[r] + bcol;
        if constexpr (HAS_EM) v *= emul[(size_t)m * FD + col];
        if constexpr (RELU)   v = fmaxf(v, 0.0f);
        size_t o = (size_t)m * FD + col;
        if constexpr (ACCUM)  out[o] += v; else out[o] = v;
    }
}

// ---------------------------------------------------------------------------
// Setup: is_t / h0 / t0 / r0 per batch, gather rel_emb.
// meta layout: [h0 x4][t0 x4][r0 x4][is_t x4]
// ---------------------------------------------------------------------------
__global__ __launch_bounds__(128)
void setup_kernel(const int* __restrict__ h_index, const int* __restrict__ r_index,
                  const int* __restrict__ t_index, const float* __restrict__ rel_table,
                  int* __restrict__ meta, float* __restrict__ rel_emb)
{
    int tid = threadIdx.x;
    if (tid < 4) {
        int b = tid;
        int h00 = h_index[b * 32];
        int ist = 1;
        for (int j = 1; j < 32; ++j)
            if (h_index[b * 32 + j] != h00) { ist = 0; break; }
        int h0 = ist ? h_index[b * 32] : t_index[b * 32];
        int t0 = ist ? t_index[b * 32] : h_index[b * 32];
        int r0 = ist ? r_index[b * 32] : r_index[b * 32] + 18;
        meta[b]      = h0 + b * NPB;
        meta[4 + b]  = t0 + b * NPB;
        meta[8 + b]  = r0;
        meta[12 + b] = ist;
    }
    __syncthreads();
    for (int b = 0; b < 4; ++b) {
        int r0 = meta[8 + b];
        rel_emb[b * FD + tid] = rel_table[r0 * FD + tid];
    }
}

__global__ void init_cur_kernel(const float* __restrict__ hidden,
                                const float* __restrict__ rel_hidden,
                                const int* __restrict__ meta,
                                float* __restrict__ cur)
{
    int t = blockIdx.x * blockDim.x + threadIdx.x;
    if (t >= NN * FD) return;
    int n = t >> 7, f = t & 127;
    float val = 0.f;
    #pragma unroll
    for (int b = 0; b < 4; ++b) if (n == meta[4 + b]) val = rel_hidden[b * FD + f];
    #pragma unroll
    for (int b = 0; b < 4; ++b) if (n == meta[b])     val = hidden[b * FD + f];
    cur[t] = val;
}

// score MLP for the 4 seed rows (tiny; scalar fp32)
__global__ __launch_bounds__(128)
void s0_kernel(const float* __restrict__ hidden, const float* __restrict__ rel_emb,
               const float* __restrict__ Wl, const float* __restrict__ bl,
               const float* __restrict__ W1, const float* __restrict__ b1,
               const float* __restrict__ W2, const float* __restrict__ b2,
               float* __restrict__ s0val)
{
    int b = blockIdx.x, f = threadIdx.x;
    __shared__ float hb[128], rb[128], zz[128], red[128];
    hb[f] = hidden[b * FD + f];
    rb[f] = rel_emb[b * FD + f];
    __syncthreads();
    float acc = bl[f];
    for (int k = 0; k < 128; ++k) acc += hb[k] * Wl[k * FD + f];
    for (int k = 0; k < 128; ++k) acc += rb[k] * Wl[(128 + k) * FD + f];
    zz[f] = acc * hb[f];
    __syncthreads();
    float acc2 = b1[f];
    for (int k = 0; k < 128; ++k) acc2 += zz[k] * W1[k * FD + f];
    red[f] = fmaxf(acc2, 0.f) * W2[f];
    __syncthreads();
    for (int s = 64; s > 0; s >>= 1) { if (f < s) red[f] += red[f + s]; __syncthreads(); }
    if (f == 0) s0val[b] = red[0] + b2[0];
}

// Layer-0 node selection: top 1.6M of the 16M-entry repeat(s0, N) array.
// Zeros/positives at indices < 8000 always selected; negatives never.
// s0 can only be nonzero at j/2000 in {0..3} when meta[0] (=h0 of batch 0) < 4.
__global__ void sel0_kernel(const int* __restrict__ meta, const float* __restrict__ s0val,
                            int* __restrict__ node_sel)
{
    int t = blockIdx.x * blockDim.x + threadIdx.x;
    if (t >= NN) return;
    int c = t / NPB;
    float v = (meta[0] == c) ? s0val[0] : 0.f;
    node_sel[t] = (v >= 0.f) ? 1 : 0;
}

// Exact top-k (k=800 of 8000), jax tie-breaking: value desc, index asc.
// Single-workgroup bitonic sort over 8192 (val,idx) pairs in 64KB LDS.
__global__ __launch_bounds__(1024)
void topk_kernel(const float* __restrict__ score, int* __restrict__ node_sel)
{
    __shared__ float sv[NSORT];
    __shared__ int   si[NSORT];
    int tid = threadIdx.x;
    for (int i = tid; i < NSORT; i += 1024) {
        sv[i] = (i < NN) ? score[i] : -3.4e38f;
        si[i] = i;
    }
    __syncthreads();
    for (int kk = 2; kk <= NSORT; kk <<= 1) {
        for (int j = kk >> 1; j > 0; j >>= 1) {
            for (int t = tid; t < NSORT; t += 1024) {
                int ixj = t ^ j;
                if (ixj > t) {
                    float va = sv[t], vb = sv[ixj];
                    int   ia = si[t], ib = si[ixj];
                    bool aAfterB = (va < vb) || (va == vb && ia > ib);
                    bool doSwap  = ((t & kk) == 0) ? aAfterB : !aAfterB;
                    if (doSwap) { sv[t] = vb; sv[ixj] = va; si[t] = ib; si[ixj] = ia; }
                }
            }
            __syncthreads();
        }
    }
    for (int i = tid; i < NN; i += 1024) node_sel[i] = 0;
    __syncthreads();
    for (int p = tid; p < KTOP; p += 1024) {
        int idx = si[p];
        if (idx < NN) node_sel[idx] = 1;
    }
}

__global__ void init_agg_kernel(float* __restrict__ Sv, float* __restrict__ Sv2,
                                float* __restrict__ Mv, float* __restrict__ mv,
                                float* __restrict__ cnt)
{
    int t = blockIdx.x * blockDim.x + threadIdx.x;
    if (t >= NN * FD) return;
    Sv[t] = 0.f; Sv2[t] = 0.f; Mv[t] = -3.0e38f; mv[t] = 3.0e38f;
    if ((t & 127) == 0) cnt[t >> 7] = 0.f;
}

// Edge reduction: per selected edge, fold v[row] into destination aggregates.
// Bandwidth/atomic bound; u/v/aggregate arrays are L2-resident.
__global__ void edge_reduce_kernel(const int* __restrict__ eidx,
                                   const float* __restrict__ v,
                                   const int* __restrict__ node_sel,
                                   float* __restrict__ Sv, float* __restrict__ Sv2,
                                   float* __restrict__ Mv, float* __restrict__ mv,
                                   float* __restrict__ cnt)
{
    long long t = (long long)blockIdx.x * blockDim.x + threadIdx.x;
    if (t >= (long long)ET * FD) return;
    int e = (int)(t >> 7), f = (int)(t & 127);
    int b  = e / NE, ee = e - b * NE;
    int row = eidx[ee] + b * NPB;
    if (!node_sel[row]) return;
    int col = eidx[NE + ee] + b * NPB;
    float vr = v[(size_t)row * FD + f];
    atomicAdd(&Sv [(size_t)col * FD + f], vr);
    atomicAdd(&Sv2[(size_t)col * FD + f], vr * vr);
    atomicMaxF(&Mv[(size_t)col * FD + f], vr);
    atomicMinF(&mv[(size_t)col * FD + f], vr);
    if (f == 0) atomicAdd(&cnt[col], 1.0f);
}

// Per-node PNA aggregate reconstruction:
//  mean  = (cnt*ub + Sv)/den,  mean2 = (cnt*ub^2 + 2 ub Sv + Sv2)/den
//  max   = has ? ub + Mv : 0,  min   = has ? ub + mv : 0,  ub = u + bpre
__global__ void agg_kernel(const float* __restrict__ u, const float* __restrict__ bpre_l,
                           const float* __restrict__ Sv, const float* __restrict__ Sv2,
                           const float* __restrict__ Mv, const float* __restrict__ mv,
                           const float* __restrict__ cnt,
                           float* __restrict__ agg, float* __restrict__ s1,
                           float* __restrict__ s2)
{
    const float AVG = 3.4965075614664802f; // log(33.0)
    int t = blockIdx.x * blockDim.x + threadIdx.x;
    if (t >= NN * FD) return;
    int n = t >> 7, f = t & 127;
    float cn  = cnt[n];
    float den = fmaxf(cn, 1.f);
    float ub  = u[t] + bpre_l[f];
    float sv  = Sv[t], sv2 = Sv2[t];
    float mean  = (cn * ub + sv) / den;
    float mean2 = (cn * ub * ub + 2.f * ub * sv + sv2) / den;
    float sd    = sqrtf(fmaxf(mean2 - mean * mean, 0.f) + 1e-5f);
    bool  has   = cn > 0.f;
    float mx = has ? ub + Mv[t] : 0.f;
    float mn = has ? ub + mv[t] : 0.f;
    size_t o = (size_t)n * 512 + f;
    agg[o] = mean; agg[o + 128] = mx; agg[o + 256] = mn; agg[o + 384] = sd;
    if (f == 0) {
        float logd = logf(den + 1.f);
        s1[n] = logd / AVG;
        s2[n] = AVG / logd;
    }
}

// score[n] = dot(z2[n], W2) + b2
__global__ __launch_bounds__(128)
void score_kernel(const float* __restrict__ z2b, const float* __restrict__ W2,
                  const float* __restrict__ b2, float* __restrict__ score)
{
    int n = blockIdx.x, f = threadIdx.x;
    __shared__ float red[128];
    red[f] = z2b[(size_t)n * FD + f] * W2[f];
    __syncthreads();
    for (int s = 64; s > 0; s >>= 1) { if (f < s) red[f] += red[f + s]; __syncthreads(); }
    if (f == 0) score[n] = red[0] + b2[0];
}

__global__ __launch_bounds__(128)
void gather_kernel(const int* __restrict__ h_index, const int* __restrict__ t_index,
                   const int* __restrict__ meta, const float* __restrict__ score,
                   float* __restrict__ out)
{
    int t = threadIdx.x;           // 128 = B * NUM_NEG
    int b = t >> 5, j = t & 31;
    int ist = meta[12 + b];
    int tg  = (ist ? t_index[b * 32 + j] : h_index[b * 32 + j]) + b * NPB;
    out[t] = score[tg];
}

// ---------------------------------------------------------------------------
extern "C" void kernel_launch(void* const* d_in, const int* in_sizes, int n_in,
                              void* d_out, int out_size, void* d_ws, size_t ws_size,
                              hipStream_t stream)
{
    (void)in_sizes; (void)n_in; (void)out_size; (void)ws_size;
    const int*   h_index    = (const int*)  d_in[0];
    const int*   r_index    = (const int*)  d_in[1];
    const int*   t_index    = (const int*)  d_in[2];
    const float* hidden     = (const float*)d_in[3];
    const float* rel_hidden = (const float*)d_in[4];
    const int*   edge_index = (const int*)  d_in[5];
    // d_in[6] (x) and d_in[7] (batch) are unused by the reference math
    const float* rel_table  = (const float*)d_in[8];
    const float* Wl   = (const float*)d_in[9];
    const float* bl   = (const float*)d_in[10];
    const float* W1   = (const float*)d_in[11];
    const float* b1   = (const float*)d_in[12];
    const float* W2   = (const float*)d_in[13];
    const float* b2   = (const float*)d_in[14];
    const float* Wpre = (const float*)d_in[15];
    const float* bpre = (const float*)d_in[16];
    const float* Wpost= (const float*)d_in[17];
    const float* bpost= (const float*)d_in[18];
    const float* Wlin = (const float*)d_in[19];
    const float* blin = (const float*)d_in[20];
    float* out = (float*)d_out;

    // Workspace carve-out (~45 MB; L2-resident on MI455X).
    char* base = (char*)d_ws; size_t off = 0;
    auto alloc = [&](size_t bytes) -> void* {
        void* p = base + off; off += (bytes + 255) & ~size_t(255); return p;
    };
    const size_t NF = (size_t)NN * FD * sizeof(float);
    float*    cur  = (float*)alloc(NF);
    float*    u    = (float*)alloc(NF);
    float*    v    = (float*)alloc(NF);
    float*    Sv   = (float*)alloc(NF);
    float*    Sv2  = (float*)alloc(NF);
    float*    Mv   = (float*)alloc(NF);
    float*    mv_  = (float*)alloc(NF);
    float*    agg  = (float*)alloc((size_t)NN * 512 * sizeof(float));
    _Float16* wpk  = (_Float16*)alloc((size_t)512 * FD * sizeof(_Float16));
    float*    cnt  = (float*)alloc(NN * sizeof(float));
    float*    scr  = (float*)alloc(NN * sizeof(float));
    float*    s1   = (float*)alloc(NN * sizeof(float));
    float*    s2   = (float*)alloc(NN * sizeof(float));
    float*    rel_emb = (float*)alloc(4 * FD * sizeof(float));
    int*      nsel = (int*)alloc(NN * sizeof(int));
    int*      meta = (int*)alloc(16 * sizeof(int));
    float*    s0v  = (float*)alloc(4 * sizeof(float));
    // Buffer reuse after aggregates are consumed each layer:
    float* hb  = Sv;    // post-GEMM hidden
    float* zb  = Sv2;   // heur*hidden
    float* z2b = Mv;    // relu(z@W1+b1)

    const dim3 GG(NN / 16), GB(256);
    auto gemm = [&](const float* A0, int K0, const float* A1, int a1div, int K,
                    const float* W, const float* bias, const float* rs,
                    const float* em, float* o, bool relu, bool accum) {
        pack_w_kernel<<<dim3((K * FD + 255) / 256), dim3(256), 0, stream>>>(W, wpk, K);
        if (A1)       gemm_wmma_kernel<true,false,true,false,false>
                        <<<GG, GB, 0, stream>>>(A0,K0,A1,a1div,K,wpk,bias,rs,em,o);
        else if (rs)  gemm_wmma_kernel<false,true,false,false,true>
                        <<<GG, GB, 0, stream>>>(A0,K0,A1,1,K,wpk,bias,rs,em,o);
        else if (relu)gemm_wmma_kernel<false,false,false,true,false>
                        <<<GG, GB, 0, stream>>>(A0,K0,A1,1,K,wpk,bias,rs,em,o);
        else if (accum)gemm_wmma_kernel<false,false,false,false,true>
                        <<<GG, GB, 0, stream>>>(A0,K0,A1,1,K,wpk,bias,rs,em,o);
        else          gemm_wmma_kernel<false,false,false,false,false>
                        <<<GG, GB, 0, stream>>>(A0,K0,A1,1,K,wpk,bias,rs,em,o);
    };

    const int TPB = 256;
    const int NF_BLK = (NN * FD + TPB - 1) / TPB;

    setup_kernel<<<1, 128, 0, stream>>>(h_index, r_index, t_index, rel_table, meta, rel_emb);
    init_cur_kernel<<<NF_BLK, TPB, 0, stream>>>(hidden, rel_hidden, meta, cur);
    s0_kernel<<<4, 128, 0, stream>>>(hidden, rel_emb, Wl, bl, W1, b1, W2, b2, s0v);

    for (int l = 0; l < 3; ++l) {
        const float* Wpre_l  = Wpre  + (size_t)l * 256 * FD;
        const float* bpre_l  = bpre  + (size_t)l * FD;
        const float* Wpost_l = Wpost + (size_t)l * 1664 * FD;
        const float* bpost_l = bpost + (size_t)l * FD;
        const float* Wlin_l  = Wlin  + (size_t)l * FD * FD;
        const float* blin_l  = blin  + (size_t)l * FD;

        // 1) node selection mask
        if (l == 0)
            sel0_kernel<<<(NN + TPB - 1) / TPB, TPB, 0, stream>>>(meta, s0v, nsel);
        else
            topk_kernel<<<1, 1024, 0, stream>>>(scr, nsel);

        // 2) u = cur @ Wpre[:128], v = cur @ Wpre[128:]   (bpre folded later)
        gemm(cur, FD, nullptr, 1, FD, Wpre_l,            nullptr, nullptr, nullptr, u, false, false);
        gemm(cur, FD, nullptr, 1, FD, Wpre_l + 128 * FD, nullptr, nullptr, nullptr, v, false, false);

        // 3-4) aggregate v[row] into per-destination stats
        init_agg_kernel<<<NF_BLK, TPB, 0, stream>>>(Sv, Sv2, Mv, mv_, cnt);
        edge_reduce_kernel<<<(ET * FD) / TPB, TPB, 0, stream>>>(
            edge_index, v, nsel, Sv, Sv2, Mv, mv_, cnt);

        // 5) reconstruct [mean|max|min|std] and degree scalers
        agg_kernel<<<NF_BLK, TPB, 0, stream>>>(u, bpre_l, Sv, Sv2, Mv, mv_, cnt, agg, s1, s2);

        // 6) h = [cur, agg, agg*s1, agg*s2] @ Wpost + bpost  (4 WMMA passes)
        gemm(cur, FD,  nullptr, 1, FD,  Wpost_l,             bpost_l, nullptr, nullptr, hb, false, false);
        gemm(agg, 512, nullptr, 1, 512, Wpost_l + 128  * FD, nullptr, nullptr, nullptr, hb, false, true);
        gemm(agg, 512, nullptr, 1, 512, Wpost_l + 640  * FD, nullptr, s1,      nullptr, hb, false, true);
        gemm(agg, 512, nullptr, 1, 512, Wpost_l + 1152 * FD, nullptr, s2,      nullptr, hb, false, true);

        // 7) cur += h @ Wlin + blin
        gemm(hb, FD, nullptr, 1, FD, Wlin_l, blin_l, nullptr, nullptr, cur, false, true);

        // 8) score = MLP(cur, rel_exp)
        gemm(cur, FD, rel_emb, NPB, 256, Wl, bl, nullptr, cur, zb, false, false); // z = heur*hidden
        gemm(zb,  FD, nullptr, 1,   FD,  W1, b1, nullptr, nullptr, z2b, true, false);
        score_kernel<<<NN, 128, 0, stream>>>(z2b, W2, b2, scr);
    }

    gather_kernel<<<1, 128, 0, stream>>>(h_index, t_index, meta, scr, out);
}
